// SphericalFourierNeuralOperatorNet_52896817218096
// MI455X (gfx1250) — matrix operand order repaired and checked by reading, not compiled, not run on hardware
//
#include <hip/hip_runtime.h>
#include <math.h>

typedef _Float16 h16;
typedef __attribute__((ext_vector_type(16))) _Float16 v16h;
typedef __attribute__((ext_vector_type(8)))  _Float16 v8h;
typedef __attribute__((ext_vector_type(8)))  float    v8f;

#define P_TOT 65160L          // 181*360 pixels
#define P_PAD 65216L          // 1019*64
#define NT_P  1019            // N tiles of 64 covering P

#ifndef M_PI
#define M_PI 3.14159265358979323846
#endif

__device__ __forceinline__ float gelu_f(float x){
  return 0.5f * x * (1.0f + erff(x * 0.70710678118654752f));
}

// ---------------------------------------------------------------------------
// Generic WMMA GEMM: D[m,n] = sum_k A[m,k] * Bt[n,k]   (f16 in, f32 acc)
// A: f16 row-major [Mpad][lda] (Mpad multiple of 32), Bt: f16 [Npad][ldb]
// (B transposed, Npad multiple of 64), out f32 [m*ldc + n] with m<M, n<N
// guards. One wave = 32(M) x 64(N) tile: 8 WMMAs per K-step, B fragments
// reused across two A rows (doubles FLOP/byte vs 16x64).
// ---------------------------------------------------------------------------
template<bool BIAS, bool GELU, bool RES>
__global__ __launch_bounds__(32)
void wmma_gemm(const h16* __restrict__ A, int lda, long sA,
               const h16* __restrict__ Bt, int ldb, long sB,
               float* __restrict__ Out, long ldc, long sC,
               const float* __restrict__ bias, const float* __restrict__ res,
               int M, int N, int K)
{
  const int lane = threadIdx.x & 31;
  const int mt = blockIdx.x, nt = blockIdx.y, z = blockIdx.z;
  const int h = lane >> 4, lm = lane & 15;

  const h16* arow0 = A + (size_t)z * sA + (size_t)(mt * 32 + lm) * lda;
  const h16* arow1 = arow0 + (size_t)16 * lda;
  const h16* brow  = Bt + (size_t)z * sB + (size_t)(nt * 64 + lm) * ldb + h * 16;
  const size_t bstep = (size_t)16 * ldb;
  float* Cb = Out + (size_t)z * sC;
  const float* Rb = RES ? (res + (size_t)z * sC) : nullptr;

  v8f accs[2][4] = {};
  for (int kk = 0; kk < K; kk += 32) {
    // A fragments (16x32): elems e <- K = kk + h*16 + e
    v8h a0lo = *(const v8h*)(arow0 + kk + h * 16);
    v8h a0hi = *(const v8h*)(arow0 + kk + 8 + h * 16);
    v8h a1lo = *(const v8h*)(arow1 + kk + h * 16);
    v8h a1hi = *(const v8h*)(arow1 + kk + 8 + h * 16);
    v16h a0, a1;
#pragma unroll
    for (int e = 0; e < 8; ++e) {
      a0[e] = a0lo[e]; a0[e + 8] = a0hi[e];
      a1[e] = a1lo[e]; a1[e + 8] = a1hi[e];
    }
    // B fragments (32x16 each): lane holds column n, elems e <- K = kk + h*16 + e
#pragma unroll
    for (int f = 0; f < 4; ++f) {
      v16h b = *(const v16h*)(brow + (size_t)f * bstep + kk);
      accs[0][f] = __builtin_amdgcn_wmma_f32_16x16x32_f16(false, a0, false, b, (short)0, accs[0][f], false, false);
      accs[1][f] = __builtin_amdgcn_wmma_f32_16x16x32_f16(false, a1, false, b, (short)0, accs[1][f], false, false);
    }
  }
  const int n0 = nt * 64 + lm;
#pragma unroll
  for (int g = 0; g < 2; ++g) {
#pragma unroll
    for (int j = 0; j < 8; ++j) {
      int m = mt * 32 + g * 16 + j + h * 8;   // C/D layout: VGPR j -> M = j + 8*(lane/16)
      if (m >= M) continue;
      float bv = BIAS ? bias[m] : 0.0f;
#pragma unroll
      for (int f = 0; f < 4; ++f) {
        int n = n0 + f * 16;
        if (n >= N) continue;
        float v = accs[g][f][j] + bv;
        if (RES) v += Rb[(size_t)m * ldc + n];
        if (GELU) v = gelu_f(v);
        Cb[(size_t)m * ldc + n] = v;
      }
    }
  }
}

// ---------------------------------------------------------------------------
// Basis generation (device; these are baked constants in the reference)
// ---------------------------------------------------------------------------
// Forward DFT  DF[mrow][lon] (384x384): rows 0..180 = cos/360, 192..372 = -sin/360
// Inverse DFT  DI[lon][mrow] (384x384): w_m*cos / -w_m*sin  (w=1 for m=0,180 else 2)
__global__ void gen_dft(h16* DF, h16* DI){
  int idx = blockIdx.x * blockDim.x + threadIdx.x;
  if (idx >= 384 * 384) return;
  int k = idx % 384, r = idx / 384;
  double c = 2.0 * M_PI / 360.0;
  float f = 0.f;
  if (k < 360) {
    if (r < 181) f = (float)(cos(c * r * k) / 360.0);
    else if (r >= 192 && r < 373) f = (float)(-sin(c * (r - 192) * k) / 360.0);
  }
  DF[idx] = (h16)f;
  float g = 0.f;
  if (r < 360) {
    if (k < 181) { float wm = (k == 0 || k == 180) ? 1.f : 2.f; g = wm * (float)cos(c * k * r); }
    else if (k >= 192 && k < 373) { int mm = k - 192; float wm = (mm == 0 || mm == 180) ? 1.f : 2.f; g = -wm * (float)sin(c * mm * r); }
  }
  DI[idx] = (h16)g;
}

// Orthonormal associated Legendre, per (m,lat) thread, 3-term recurrence in l.
// PW[m][l][lat] includes 2*pi*wq quadrature weight (analysis A), PS[m][lat][l] raw (synthesis A).
__global__ void gen_legendre(h16* PW, h16* PS){
  int idx = blockIdx.x * blockDim.x + threadIdx.x;
  if (idx >= 181 * 192) return;
  int lat = idx % 192, m = idx / 192;
  if (lat >= 181) {
    for (int l = 0; l < 192; ++l) {
      PW[((size_t)m * 192 + l) * 192 + lat] = (h16)0.0f;
      PS[((size_t)m * 192 + lat) * 192 + l] = (h16)0.0f;
    }
    return;
  }
  double theta = M_PI * (double)lat / 180.0;
  double ct = cos(theta), st = sin(theta);
  double w = 2.0 * M_PI * (M_PI / 180.0) * st;
  double pmm = sqrt(1.0 / (4.0 * M_PI));
  for (int j = 1; j <= m; ++j) pmm *= -sqrt((2.0 * j + 1.0) / (2.0 * j)) * st;
  double pl2 = 0.0, pl1 = 0.0;
  for (int l = 0; l < 192; ++l) {
    double v;
    if (l < m || l > 180) v = 0.0;
    else if (l == m) v = pmm;
    else if (l == m + 1) v = sqrt(2.0 * m + 3.0) * ct * pmm;
    else {
      double ll = (double)l * l, mm2 = (double)m * m;
      double a = sqrt((4.0 * ll - 1.0) / (ll - mm2));
      double b = sqrt(((2.0 * l + 1.0) * (l - 1.0 + m) * (l - 1.0 - m)) / ((2.0 * l - 3.0) * (ll - mm2)));
      v = a * ct * pl1 - b * pl2;
    }
    pl2 = pl1; pl1 = v;
    PW[((size_t)m * 192 + l) * 192 + lat] = (h16)(v * w);
    PS[((size_t)m * 192 + lat) * 192 + l] = (h16)v;
  }
}

// ---------------------------------------------------------------------------
// Pack / transpose / convert kernels (all zero-fill their pads)
// ---------------------------------------------------------------------------
__global__ void pack_a(const float* src, int rs, int M, int K, h16* dst, int Kpad, int total){
  int idx = blockIdx.x * blockDim.x + threadIdx.x;
  if (idx >= total) return;
  int m = idx / Kpad, k = idx - m * Kpad;
  dst[idx] = (m < M && k < K) ? (h16)src[(size_t)m * rs + k] : (h16)0.0f;
}

// activations [C][P] f32 (channel-major) -> Bt f16 [Ppad][Kpad]
__global__ void pack_bt(const float* src, int C, long Pl, int Kpad, long Ppad, h16* dst){
  size_t idx = (size_t)blockIdx.x * blockDim.x + threadIdx.x;
  if (idx >= (size_t)Ppad * Kpad) return;
  long p = (long)(idx / Kpad); int k = (int)(idx - (size_t)p * Kpad);
  dst[idx] = (p < Pl && k < C) ? (h16)src[(size_t)k * Pl + p] : (h16)0.0f;
}

// big-skip concat: [h(256) ; x(26)] -> Bt [Ppad][288]
__global__ void pack_btdec(const float* hbuf, const float* xin, long Pl, long Ppad, h16* dst){
  size_t idx = (size_t)blockIdx.x * blockDim.x + threadIdx.x;
  if (idx >= (size_t)Ppad * 288) return;
  long p = (long)(idx / 288); int k = (int)(idx - (size_t)p * 288);
  float v = 0.f;
  if (p < Pl) {
    if (k < 256) v = hbuf[(size_t)k * Pl + p];
    else if (k < 282) v = xin[(size_t)(k - 256) * Pl + p];
  }
  dst[idx] = (h16)v;
}

// xn [256][P] -> Xt f16 [181][256][384]  (Bt for forward DFT, per-lat batch)
__global__ void pack_xt(const float* xn, h16* dst){
  size_t idx = (size_t)blockIdx.x * blockDim.x + threadIdx.x;
  if (idx >= (size_t)181 * 256 * 384) return;
  int lon = (int)(idx % 384); size_t t = idx / 384; int c = (int)(t % 256); int lat = (int)(t / 256);
  dst[idx] = (lon < 360) ? (h16)xn[(size_t)c * P_TOT + (size_t)lat * 360 + lon] : (h16)0.0f;
}

// F f32 [181lat][384mrow][256c] -> Fb f16 [384mrow][256c][192lat]  (Bt for Legendre analysis)
__global__ void pack_f(const float* F, h16* dst){
  size_t idx = (size_t)blockIdx.x * blockDim.x + threadIdx.x;
  if (idx >= (size_t)384 * 256 * 192) return;
  int lat = (int)(idx % 192); size_t t = idx / 192; int c = (int)(t % 256); int mr = (int)(t / 256);
  dst[idx] = (lat < 181) ? (h16)F[((size_t)lat * 384 + mr) * 256 + c] : (h16)0.0f;
}

// coeff f32 [181m][192l][256c] -> Bsp f16 [181l][192m][256c]  (Bt for spectral conv)
__global__ void pack_coeff(const float* src, h16* dst){
  size_t idx = (size_t)blockIdx.x * blockDim.x + threadIdx.x;
  if (idx >= (size_t)181 * 192 * 256) return;
  int i = (int)(idx % 256); size_t t = idx / 256; int m = (int)(t % 192); int l = (int)(t / 192);
  dst[idx] = (m < 181) ? (h16)src[((size_t)m * 192 + l) * 256 + i] : (h16)0.0f;
}

// co f32 [181l][256o][192m] -> Bsyn f16 [181m][256o][192l]  (Bt for synthesis)
__global__ void pack_co(const float* src, h16* dst){
  size_t idx = (size_t)blockIdx.x * blockDim.x + threadIdx.x;
  if (idx >= (size_t)181 * 256 * 192) return;
  int l = (int)(idx % 192); size_t t = idx / 192; int o = (int)(t % 256); int m = (int)(t / 256);
  dst[idx] = (l < 181) ? (h16)src[((size_t)l * 256 + o) * 192 + m] : (h16)0.0f;
}

// Gre/Gim f32 [181m][192lat][256o] -> Gb f16 [181lat][256o][384mrow] (Bt for inverse DFT)
__global__ void pack_g(const float* gre, const float* gim, h16* dst){
  size_t idx = (size_t)blockIdx.x * blockDim.x + threadIdx.x;
  if (idx >= (size_t)181 * 256 * 384) return;
  int mr = (int)(idx % 384); size_t t = idx / 384; int o = (int)(t % 256); int lat = (int)(t / 256);
  float v = 0.f;
  if (mr < 181) v = gre[((size_t)mr * 192 + lat) * 256 + o];
  else if (mr >= 192 && mr < 373) v = gim[((size_t)(mr - 192) * 192 + lat) * 256 + o];
  dst[idx] = (h16)v;
}

// w_spec[layer] f32 [o][i][181l] -> AW f16 [181l][256o][256i]
__global__ void pack_wspec(const float* src, h16* dst){
  size_t idx = (size_t)blockIdx.x * blockDim.x + threadIdx.x;
  if (idx >= (size_t)181 * 256 * 256) return;
  int i = (int)(idx % 256); size_t t = idx / 256; int o = (int)(t % 256); int l = (int)(t / 256);
  dst[idx] = (h16)src[((size_t)o * 256 + i) * 181 + l];
}

// ---------------------------------------------------------------------------
// Elementwise kernels
// ---------------------------------------------------------------------------
__global__ void ln_kernel(const float* in, float* out, int C, long Pl){
  long p = (long)blockIdx.x * blockDim.x + threadIdx.x;
  if (p >= Pl) return;
  float s = 0.f, s2 = 0.f;
  for (int c = 0; c < C; ++c) { float v = in[(size_t)c * Pl + p]; s += v; s2 += v * v; }
  float mu = s / C;
  float var = s2 / C - mu * mu;
  float r = rsqrtf(var + 1e-6f);
  for (int c = 0; c < C; ++c) { size_t i = (size_t)c * Pl + p; out[i] = (in[i] - mu) * r; }
}

// x1 = gelu(conv_is_out + isht_out + b_spec);  ytmp laid out [P][256]
__global__ void fuse_x1(float* x1, const float* ytmp, const float* bsp, long Pl){
  size_t idx = (size_t)blockIdx.x * blockDim.x + threadIdx.x;
  if (idx >= (size_t)256 * Pl) return;
  int c = (int)(idx / Pl); size_t p = idx - (size_t)c * Pl;
  float v = x1[idx] + ytmp[p * 256 + c] + bsp[c];
  x1[idx] = gelu_f(v);
}

// ---------------------------------------------------------------------------
// Host side
// ---------------------------------------------------------------------------
static inline void gemm(hipStream_t s, int flags,
                        const h16* A, int lda, long sA,
                        const h16* B, int ldb, long sB,
                        float* C, long ldc, long sC,
                        const float* bias, const float* res,
                        int M, int Mt /* tiles of 32 */, int N, int Nt /* tiles of 64 */,
                        int K, int batch)
{
  dim3 g(Mt, Nt, batch), b(32, 1, 1);
  if (flags == 0)
    wmma_gemm<false, false, false><<<g, b, 0, s>>>(A, lda, sA, B, ldb, sB, C, ldc, sC, bias, res, M, N, K);
  else if (flags == 1)
    wmma_gemm<true, false, false><<<g, b, 0, s>>>(A, lda, sA, B, ldb, sB, C, ldc, sC, bias, res, M, N, K);
  else if (flags == 3)
    wmma_gemm<true, true, false><<<g, b, 0, s>>>(A, lda, sA, B, ldb, sB, C, ldc, sC, bias, res, M, N, K);
  else
    wmma_gemm<true, false, true><<<g, b, 0, s>>>(A, lda, sA, B, ldb, sB, C, ldc, sC, bias, res, M, N, K);
}

#define GRID1(n) dim3((unsigned)(((n) + 255) / 256)), dim3(256)

extern "C" void kernel_launch(void* const* d_in, const int* in_sizes, int n_in,
                              void* d_out, int out_size, void* d_ws, size_t ws_size,
                              hipStream_t stream) {
  (void)in_sizes; (void)n_in; (void)out_size; (void)ws_size;
  const float* x      = (const float*)d_in[0];   // [26][65160]
  const float* w_enc  = (const float*)d_in[1];   // [256][26]
  const float* b_enc  = (const float*)d_in[2];
  const float* posemb = (const float*)d_in[3];   // [256][65160]
  const float* w_spec = (const float*)d_in[4];   // [4][256][256][181]
  const float* b_spec = (const float*)d_in[5];
  const float* w_isk  = (const float*)d_in[6];   // [4][256][256]
  const float* b_isk  = (const float*)d_in[7];
  const float* w_fc1  = (const float*)d_in[8];   // [4][512][256]
  const float* b_fc1  = (const float*)d_in[9];
  const float* w_fc2  = (const float*)d_in[10];  // [4][256][512]
  const float* b_fc2  = (const float*)d_in[11];
  const float* w_dec  = (const float*)d_in[12];  // [26][282]
  const float* b_dec  = (const float*)d_in[13];
  float* outp = (float*)d_out;                    // [26][65160]

  char* base = (char*)d_ws;
  size_t off = 0;
  auto alloc = [&](size_t elems, size_t esize) -> void* {
    void* p = base + off;
    off += (elems * esize + 255) & ~(size_t)255;
    return p;
  };
  float* H   = (float*)alloc(256 * (size_t)P_TOT, 4);
  float* XN  = (float*)alloc(256 * (size_t)P_TOT, 4);
  float* X1  = (float*)alloc(256 * (size_t)P_TOT, 4);
  float* X2  = (float*)alloc(256 * (size_t)P_TOT, 4);
  float* H1  = (float*)alloc(512 * (size_t)P_TOT, 4);
  float* R1  = (float*)alloc(17793024, 4);   // F | co(re+im) | Ytmp
  float* R2  = (float*)alloc(17793024, 4);   // coeff(re+im) | G(re+im)
  h16* BT    = (h16*)alloc((size_t)P_PAD * 512, 2);
  h16* S1    = (h16*)alloc(17793024, 2);     // Xt | Bsp(re+im) | Gb
  h16* S2    = (h16*)alloc(18874368, 2);     // Fb | Bsyn(re+im)
  h16* AW    = (h16*)alloc(11862016, 2);     // packed w_spec[layer]
  h16* PWB   = (h16*)alloc(6672384, 2);
  h16* PSB   = (h16*)alloc(6672384, 2);
  h16* DFTF  = (h16*)alloc(147456, 2);
  h16* DFTI  = (h16*)alloc(147456, 2);
  h16* AENC  = (h16*)alloc(8192, 2);
  h16* AWIS  = (h16*)alloc(65536, 2);
  h16* AWF1  = (h16*)alloc(131072, 2);
  h16* AWF2  = (h16*)alloc(131072, 2);
  h16* ADEC  = (h16*)alloc(9216, 2);

  const long HALF = 8896512;        // 181*192*256 == 181*256*192

  // ---- constants & static weight packs ----
  gen_dft<<<GRID1(147456), 0, stream>>>(DFTF, DFTI);
  gen_legendre<<<GRID1(181 * 192), 0, stream>>>(PWB, PSB);
  pack_a<<<GRID1(256 * 32), 0, stream>>>(w_enc, 26, 256, 26, AENC, 32, 256 * 32);
  pack_a<<<GRID1(32 * 288), 0, stream>>>(w_dec, 282, 26, 282, ADEC, 288, 32 * 288);

  // ---- encoder: H = w_enc @ x + b_enc + pos_emb ----
  pack_bt<<<GRID1(P_PAD * 32), 0, stream>>>(x, 26, P_TOT, 32, P_PAD, BT);
  gemm(stream, 5, AENC, 32, 0, BT, 32, 0, H, P_TOT, 0, b_enc, posemb, 256, 8, (int)P_TOT, NT_P, 32, 1);

  for (int L = 0; L < 4; ++L) {
    const float* wsp = w_spec + (size_t)L * 256 * 256 * 181;
    const float* bsp = b_spec + (size_t)L * 256;
    const float* wis = w_isk  + (size_t)L * 256 * 256;
    const float* bis = b_isk  + (size_t)L * 256;
    const float* wf1 = w_fc1  + (size_t)L * 512 * 256;
    const float* bf1 = b_fc1  + (size_t)L * 512;
    const float* wf2 = w_fc2  + (size_t)L * 256 * 512;
    const float* bf2 = b_fc2  + (size_t)L * 256;

    // xn = LN(h)
    ln_kernel<<<GRID1(P_TOT), 0, stream>>>(H, XN, 256, P_TOT);

    // ---- SHT: forward DFT over longitude (batched over 181 lats) ----
    pack_xt<<<GRID1((size_t)181 * 256 * 384), 0, stream>>>(XN, S1);
    gemm(stream, 0, DFTF, 384, 0, S1, 384, 256L * 384, R1, 256, 384L * 256,
         nullptr, nullptr, 384, 12, 256, 4, 384, 181);
    pack_f<<<GRID1((size_t)384 * 256 * 192), 0, stream>>>(R1, S2);

    // ---- Legendre analysis (batched over 181 m, re & im) ----
    gemm(stream, 0, PWB, 192, 192L * 192, S2, 192, 256L * 192, R2, 256, 192L * 256,
         nullptr, nullptr, 192, 6, 256, 4, 192, 181);
    gemm(stream, 0, PWB, 192, 192L * 192, S2 + (size_t)192 * 256 * 192, 192, 256L * 192,
         R2 + HALF, 256, 192L * 256, nullptr, nullptr, 192, 6, 256, 4, 192, 181);

    // ---- spectral dhconv (batched over 181 l, re & im) ----
    pack_wspec<<<GRID1((size_t)181 * 256 * 256), 0, stream>>>(wsp, AW);
    pack_coeff<<<GRID1((size_t)HALF), 0, stream>>>(R2, S1);
    pack_coeff<<<GRID1((size_t)HALF), 0, stream>>>(R2 + HALF, S1 + HALF);
    gemm(stream, 0, AW, 256, 256L * 256, S1, 256, 192L * 256, R1, 192, 256L * 192,
         nullptr, nullptr, 256, 8, 192, 3, 256, 181);
    gemm(stream, 0, AW, 256, 256L * 256, S1 + HALF, 256, 192L * 256, R1 + HALF, 192, 256L * 192,
         nullptr, nullptr, 256, 8, 192, 3, 256, 181);

    // ---- Legendre synthesis (batched over 181 m, re & im) ----
    pack_co<<<GRID1((size_t)HALF), 0, stream>>>(R1, S2);
    pack_co<<<GRID1((size_t)HALF), 0, stream>>>(R1 + HALF, S2 + HALF);
    gemm(stream, 0, PSB, 192, 192L * 192, S2, 192, 256L * 192, R2, 256, 192L * 256,
         nullptr, nullptr, 192, 6, 256, 4, 192, 181);
    gemm(stream, 0, PSB, 192, 192L * 192, S2 + HALF, 192, 256L * 192, R2 + HALF, 256, 192L * 256,
         nullptr, nullptr, 192, 6, 256, 4, 192, 181);

    // ---- inverse DFT over longitude -> Ytmp [P][256] in R1 ----
    pack_g<<<GRID1((size_t)181 * 256 * 384), 0, stream>>>(R2, R2 + HALF, S1);
    gemm(stream, 0, DFTI, 384, 0, S1, 384, 256L * 384, R1, 256, 360L * 256,
         nullptr, nullptr, 360, 12, 256, 4, 384, 181);

    // ---- inner skip conv1x1 on xn, then fuse + gelu ----
    pack_a<<<GRID1(65536), 0, stream>>>(wis, 256, 256, 256, AWIS, 256, 65536);
    pack_bt<<<GRID1(P_PAD * 256), 0, stream>>>(XN, 256, P_TOT, 256, P_PAD, BT);
    gemm(stream, 1, AWIS, 256, 0, BT, 256, 0, X1, P_TOT, 0, bis, nullptr,
         256, 8, (int)P_TOT, NT_P, 256, 1);
    fuse_x1<<<GRID1((size_t)256 * P_TOT), 0, stream>>>(X1, R1, bsp, P_TOT);

    // ---- MLP with outer residual (+xn) ----
    ln_kernel<<<GRID1(P_TOT), 0, stream>>>(X1, X2, 256, P_TOT);
    pack_bt<<<GRID1(P_PAD * 256), 0, stream>>>(X2, 256, P_TOT, 256, P_PAD, BT);
    pack_a<<<GRID1(131072), 0, stream>>>(wf1, 256, 512, 256, AWF1, 256, 131072);
    gemm(stream, 3, AWF1, 256, 0, BT, 256, 0, H1, P_TOT, 0, bf1, nullptr,
         512, 16, (int)P_TOT, NT_P, 256, 1);
    pack_bt<<<GRID1(P_PAD * 512), 0, stream>>>(H1, 512, P_TOT, 512, P_PAD, BT);
    pack_a<<<GRID1(131072), 0, stream>>>(wf2, 512, 256, 512, AWF2, 512, 131072);
    gemm(stream, 5, AWF2, 512, 0, BT, 512, 0, H, P_TOT, 0, bf2, XN,
         256, 8, (int)P_TOT, NT_P, 512, 1);
  }

  // ---- decoder on [h ; x] concat ----
  pack_btdec<<<GRID1(P_PAD * 288), 0, stream>>>(H, x, P_TOT, P_PAD, BT);
  gemm(stream, 1, ADEC, 288, 0, BT, 288, 0, outp, P_TOT, 0, b_dec, nullptr,
       26, 1, (int)P_TOT, NT_P, 288, 1);
}
